// Multihead_Attention_31018253812438
// MI455X (gfx1250) — matrix operand order repaired and checked by reading, not compiled, not run on hardware
//
#include <hip/hip_runtime.h>

// ---------------------------------------------------------------------------
// Types
// ---------------------------------------------------------------------------
typedef __bf16 bf16;
typedef __attribute__((ext_vector_type(16))) __bf16 bf16x16;
typedef __attribute__((ext_vector_type(8)))  __bf16 bf16x8;
typedef __attribute__((ext_vector_type(8)))  float  f32x8;

#define B_      2
#define S_      2048
#define EMB_    1024
#define HEADS_  16
#define HDIM_   64
#define NTOK_   (B_ * S_)      // 4096
#define PSTRIDE 56             // LDS row stride in bf16 (112B: 16B aligned, bank-conflict-free)

// ---------------------------------------------------------------------------
// WMMA helpers (CDNA5 wave32 layouts, CDNA5 ISA 7.12.2)
// ---------------------------------------------------------------------------
__device__ inline f32x8 wmma_bf16(bf16x16 a, bf16x16 b, f32x8 c) {
    // 8-arg form: (neg_a, A, neg_b, B, c_mod, C, reuse_a, reuse_b)
    return __builtin_amdgcn_wmma_f32_16x16x32_bf16(false, a, false, b, (short)0, c, false, false);
}

// A-matrix 16x32 bf16 fragment from row-major [*, ld] at (row0.., k0..k0+31).
// Lane L(0-15): M=L, K in {k0+0..7, k0+16..23}; Lane L+16: M=L, K in {k0+8..15, k0+24..31}.
__device__ inline bf16x16 load_afrag(const bf16* __restrict__ base, int ld,
                                     int row0, int k0, int lane) {
    int m  = lane & 15;
    int hi = lane >> 4;
    const bf16* p = base + (size_t)(row0 + m) * ld + k0 + hi * 8;
    bf16x8 lo = *(const bf16x8*)(p);          // 16B load
    bf16x8 hv = *(const bf16x8*)(p + 16);     // 16B load
    bf16x16 r;
#pragma unroll
    for (int i = 0; i < 8; i++) { r[i] = lo[i]; r[i + 8] = hv[i]; }
    return r;
}

// B-matrix 32x16 fragment where logical B[k,n] = Bsrc[n0+n][k0+k], Bsrc row-major [*, ld].
// Lane L(0-15): N=L, K in {k0+0..7, k0+16..23}; Lane L+16: N=L, K in {k0+8..15, k0+24..31}.
__device__ inline bf16x16 load_bfrag(const bf16* __restrict__ base, int ld,
                                     int n0, int k0, int lane) {
    int n  = lane & 15;
    int hi = lane >> 4;
    const bf16* p = base + (size_t)(n0 + n) * ld + k0 + hi * 8;
    bf16x8 lo = *(const bf16x8*)(p);
    bf16x8 hv = *(const bf16x8*)(p + 16);
    bf16x16 r;
#pragma unroll
    for (int i = 0; i < 8; i++) { r[i] = lo[i]; r[i + 8] = hv[i]; }
    return r;
}

// ---------------------------------------------------------------------------
// fp32 -> bf16 conversion
// ---------------------------------------------------------------------------
__global__ void f32_to_bf16_kern(const float* __restrict__ in, bf16* __restrict__ out, int n) {
    int i = blockIdx.x * blockDim.x + threadIdx.x;
    if (i < n) out[i] = (bf16)in[i];
}

// ---------------------------------------------------------------------------
// Generic TN GEMM: C[M,N] = A[M,K] * Bsrc[N,K]^T (+bias) with epilogue modes:
//   mode 0: bf16 store row-major [M,N], bias[col], scaled (Q: 1/sqrt(64), K: 1)
//   mode 1: bf16 store into Vt[B,H,D,S]  (rows = emb channel c, cols n = b*S+s), bias[row]
//   mode 2: f32  store row-major [M,N], bias[col]  (final output projection)
// Block: 256 thr = 8 waves as 4x2; wave tile 32x64 (2x4 WMMA tiles); block tile 128x128.
// Per K=32 step: 8 WMMAs vs 6 fragment loads.
// ---------------------------------------------------------------------------
__global__ __launch_bounds__(256)
void gemm_tn_kern(const bf16* __restrict__ A, const bf16* __restrict__ Bsrc,
                  const float* __restrict__ bias, void* __restrict__ out,
                  int N, int K, int mode, float scale) {
    int lane = threadIdx.x & 31;
    int wave = threadIdx.x >> 5;
    int row0 = blockIdx.x * 128 + (wave >> 1) * 32;
    int col0 = blockIdx.y * 128 + (wave &  1) * 64;

    const f32x8 zf = {0.f, 0.f, 0.f, 0.f, 0.f, 0.f, 0.f, 0.f};
    f32x8 acc[2][4];
#pragma unroll
    for (int i = 0; i < 2; i++)
#pragma unroll
        for (int j = 0; j < 4; j++) acc[i][j] = zf;

    for (int k0 = 0; k0 < K; k0 += 32) {
        bf16x16 a0 = load_afrag(A, K, row0,      k0, lane);
        bf16x16 a1 = load_afrag(A, K, row0 + 16, k0, lane);
#pragma unroll
        for (int j = 0; j < 4; j++) {
            bf16x16 bj = load_bfrag(Bsrc, K, col0 + j * 16, k0, lane);
            acc[0][j] = wmma_bf16(a0, bj, acc[0][j]);
            acc[1][j] = wmma_bf16(a1, bj, acc[1][j]);
        }
    }

    int hi = lane >> 4, nn = lane & 15;
#pragma unroll
    for (int i = 0; i < 2; i++)
#pragma unroll
        for (int j = 0; j < 4; j++) {
            int rbase = row0 + i * 16 + hi * 8;
            int col   = col0 + j * 16 + nn;
#pragma unroll
            for (int r = 0; r < 8; r++) {
                int row = rbase + r;
                float v = acc[i][j][r];
                v += (mode == 1) ? bias[row] : bias[col];
                v *= scale;
                if (mode == 2) {
                    ((float*)out)[(size_t)row * N + col] = v;
                } else if (mode == 0) {
                    ((bf16*)out)[(size_t)row * N + col] = (bf16)v;
                } else { // Vt[B,H,D,S]: flat = ((b*EMB + row)*S + s), col = b*S + s
                    int bb = col >> 11, s = col & (S_ - 1);
                    ((bf16*)out)[((size_t)bb * EMB_ + row) * S_ + s] = (bf16)v;
                }
            }
        }
}

// ---------------------------------------------------------------------------
// Flash attention: one wave per (b, h, 32-row q tile). 8 waves / block.
// Q,K: bf16 [B,S,EMB] (Q pre-scaled by 1/sqrt(D)); Vt: bf16 [B,H,D,S]; out: bf16 [B,S,EMB]
// Per 32-key step: 16 WMMAs (8 score + 8 P*V) vs 8 global fragment loads.
// ---------------------------------------------------------------------------
__global__ __launch_bounds__(256)
void attn_fa_kern(const bf16* __restrict__ Q, const bf16* __restrict__ Kb,
                  const bf16* __restrict__ Vt, bf16* __restrict__ Oattn) {
    __shared__ __attribute__((aligned(16))) bf16 lds[8][32 * PSTRIDE];

    int lane = threadIdx.x & 31;
    int wave = threadIdx.x >> 5;
    int task = blockIdx.x * 8 + wave;         // 0..2047
    int qt = task & 63;                       // 32-row q tile within sequence
    int bh = task >> 6;                       // 0..31
    int b = bh >> 4, h = bh & 15;
    int q0 = qt * 32;
    int m = lane & 15, hi = lane >> 4;

    const bf16* Qbase = Q  + (size_t)b * S_ * EMB_ + (size_t)h * HDIM_;
    const bf16* Kbase = Kb + (size_t)b * S_ * EMB_ + (size_t)h * HDIM_;
    const bf16* Vbase = Vt + (size_t)(b * HEADS_ + h) * HDIM_ * S_;
    bf16* pl = lds[wave];

    // Q fragments (32 x 64) held in registers for the whole pass
    bf16x16 qa[2][2];
#pragma unroll
    for (int qi = 0; qi < 2; qi++) {
        qa[qi][0] = load_afrag(Qbase, EMB_, q0 + qi * 16, 0,  lane);
        qa[qi][1] = load_afrag(Qbase, EMB_, q0 + qi * 16, 32, lane);
    }

    const f32x8 zf = {0.f, 0.f, 0.f, 0.f, 0.f, 0.f, 0.f, 0.f};
    f32x8 o[2][4];
    float mrow[2][8], lrow[2][8];
#pragma unroll
    for (int qi = 0; qi < 2; qi++) {
#pragma unroll
        for (int t4 = 0; t4 < 4; t4++) o[qi][t4] = zf;
#pragma unroll
        for (int r = 0; r < 8; r++) { mrow[qi][r] = -1e30f; lrow[qi][r] = 0.f; }
    }

    for (int kk = 0; kk < S_; kk += 32) {
        // K fragments loaded once, shared across both q row-tiles
        bf16x16 kb00 = load_bfrag(Kbase, EMB_, kk,      0,  lane);
        bf16x16 kb01 = load_bfrag(Kbase, EMB_, kk,      32, lane);
        bf16x16 kb10 = load_bfrag(Kbase, EMB_, kk + 16, 0,  lane);
        bf16x16 kb11 = load_bfrag(Kbase, EMB_, kk + 16, 32, lane);

        f32x8 s[2][2];
#pragma unroll
        for (int qi = 0; qi < 2; qi++) {
            s[qi][0] = wmma_bf16(qa[qi][0], kb00, zf);
            s[qi][0] = wmma_bf16(qa[qi][1], kb01, s[qi][0]);
            s[qi][1] = wmma_bf16(qa[qi][0], kb10, zf);
            s[qi][1] = wmma_bf16(qa[qi][1], kb11, s[qi][1]);
        }

        // online softmax; row (r + hi*8) lives across the 16 lanes of this half-wave
#pragma unroll
        for (int qi = 0; qi < 2; qi++) {
#pragma unroll
            for (int r = 0; r < 8; r++) {
                float t = fmaxf(s[qi][0][r], s[qi][1][r]);
                t = fmaxf(t, __shfl_xor(t, 1, 32));
                t = fmaxf(t, __shfl_xor(t, 2, 32));
                t = fmaxf(t, __shfl_xor(t, 4, 32));
                t = fmaxf(t, __shfl_xor(t, 8, 32));
                float mnew = fmaxf(mrow[qi][r], t);
                float corr = __expf(mrow[qi][r] - mnew);
                float p0 = __expf(s[qi][0][r] - mnew);
                float p1 = __expf(s[qi][1][r] - mnew);
                float ps = p0 + p1;
                ps += __shfl_xor(ps, 1, 32);
                ps += __shfl_xor(ps, 2, 32);
                ps += __shfl_xor(ps, 4, 32);
                ps += __shfl_xor(ps, 8, 32);
                lrow[qi][r] = lrow[qi][r] * corr + ps;
                mrow[qi][r] = mnew;
#pragma unroll
                for (int t4 = 0; t4 < 4; t4++) o[qi][t4][r] *= corr;
                // stage P (acc layout -> plain [row][col] in per-wave LDS)
                int rowoff = (qi * 16 + r + hi * 8) * PSTRIDE;
                pl[rowoff + m]      = (bf16)p0;
                pl[rowoff + 16 + m] = (bf16)p1;
            }
        }

        // reload P as A-fragments (2 x 16x32); V fragments loaded once, shared
        bf16x16 pa[2];
#pragma unroll
        for (int qi = 0; qi < 2; qi++) {
            const bf16* pp = pl + (qi * 16 + m) * PSTRIDE + hi * 8;
            bf16x8 lo = *(const bf16x8*)(pp);
            bf16x8 hv = *(const bf16x8*)(pp + 16);
#pragma unroll
            for (int i = 0; i < 8; i++) { pa[qi][i] = lo[i]; pa[qi][i + 8] = hv[i]; }
        }
#pragma unroll
        for (int t4 = 0; t4 < 4; t4++) {
            bf16x16 vb = load_bfrag(Vbase, S_, t4 * 16, kk, lane);
            o[0][t4] = wmma_bf16(pa[0], vb, o[0][t4]);
            o[1][t4] = wmma_bf16(pa[1], vb, o[1][t4]);
        }
    }

    // finalize: divide by row sums, store bf16 [B,S,EMB]
#pragma unroll
    for (int qi = 0; qi < 2; qi++) {
        bf16* obase = Oattn + (size_t)(b * S_ + q0 + qi * 16) * EMB_ + h * HDIM_;
#pragma unroll
        for (int r = 0; r < 8; r++) {
            float inv = 1.0f / lrow[qi][r];
#pragma unroll
            for (int t4 = 0; t4 < 4; t4++)
                obase[(size_t)(r + hi * 8) * EMB_ + t4 * 16 + m] = (bf16)(o[qi][t4][r] * inv);
        }
    }
}

// ---------------------------------------------------------------------------
// Host launch
// ---------------------------------------------------------------------------
extern "C" void kernel_launch(void* const* d_in, const int* in_sizes, int n_in,
                              void* d_out, int out_size, void* d_ws, size_t ws_size,
                              hipStream_t stream) {
    const float* x  = (const float*)d_in[0];
    const float* y  = (const float*)d_in[1];
    // d_in[2] = mask: faithfully ignored (reference discards masked_fill result)
    const float* Wq = (const float*)d_in[3];
    const float* bq = (const float*)d_in[4];
    const float* Wk = (const float*)d_in[5];
    const float* bk = (const float*)d_in[6];
    const float* Wv = (const float*)d_in[7];
    const float* bv = (const float*)d_in[8];
    const float* Wo = (const float*)d_in[9];
    const float* bo = (const float*)d_in[10];
    float* out = (float*)d_out;

    const size_t NTE = (size_t)NTOK_ * EMB_;   // 4,194,304
    const size_t WE  = (size_t)EMB_ * EMB_;    // 1,048,576

    char* p = (char*)d_ws;
    bf16* xb   = (bf16*)p; p += NTE * 2;
    bf16* yb   = (bf16*)p; p += NTE * 2;
    bf16* wqb  = (bf16*)p; p += WE * 2;
    bf16* wkb  = (bf16*)p; p += WE * 2;
    bf16* wvb  = (bf16*)p; p += WE * 2;
    bf16* wob  = (bf16*)p; p += WE * 2;
    bf16* Qb   = (bf16*)p; p += NTE * 2;
    bf16* Kbf  = (bf16*)p; p += NTE * 2;
    bf16* Vtb  = (bf16*)p; p += NTE * 2;
    bf16* atb  = (bf16*)p; p += NTE * 2;       // total 56 MB

    // 1) bf16 conversions
    f32_to_bf16_kern<<<(int)((NTE + 255) / 256), 256, 0, stream>>>(x,  xb,  (int)NTE);
    f32_to_bf16_kern<<<(int)((NTE + 255) / 256), 256, 0, stream>>>(y,  yb,  (int)NTE);
    f32_to_bf16_kern<<<(int)((WE  + 255) / 256), 256, 0, stream>>>(Wq, wqb, (int)WE);
    f32_to_bf16_kern<<<(int)((WE  + 255) / 256), 256, 0, stream>>>(Wk, wkb, (int)WE);
    f32_to_bf16_kern<<<(int)((WE  + 255) / 256), 256, 0, stream>>>(Wv, wvb, (int)WE);
    f32_to_bf16_kern<<<(int)((WE  + 255) / 256), 256, 0, stream>>>(Wo, wob, (int)WE);

    // 2) projections (block tile 128x128)
    const float qscale = 0.125f;               // 1/sqrt(HEAD_DIM=64)
    dim3 gProj(NTOK_ / 128, EMB_ / 128);       // 32 x 8
    gemm_tn_kern<<<gProj, 256, 0, stream>>>(xb, wqb, bq, Qb,  EMB_, EMB_, 0, qscale);
    gemm_tn_kern<<<gProj, 256, 0, stream>>>(yb, wkb, bk, Kbf, EMB_, EMB_, 0, 1.0f);
    // V computed transposed: rows = output channel (Wv rows), cols = tokens
    dim3 gVt(EMB_ / 128, NTOK_ / 128);         // 8 x 32
    gemm_tn_kern<<<gVt, 256, 0, stream>>>(wvb, yb, bv, Vtb, NTOK_, EMB_, 1, 1.0f);

    // 3) fused flash attention (2048 wave tasks, 8 waves/block)
    attn_fa_kern<<<256, 256, 0, stream>>>(Qb, Kbf, Vtb, atb);

    // 4) output projection -> fp32 d_out
    gemm_tn_kern<<<gProj, 256, 0, stream>>>(atb, wob, bo, out, EMB_, EMB_, 2, 1.0f);
}
